// CTMR_UncFusionNet_11536282157662
// MI455X (gfx1250) — matrix operand order repaired and checked by reading, not compiled, not run on hardware
//
#include <hip/hip_runtime.h>
#include <math.h>

typedef __attribute__((ext_vector_type(16))) __bf16 v16bf;
typedef __attribute__((ext_vector_type(8)))  float  v8f;
typedef unsigned short u16;
typedef unsigned long long u64;

#define BATCH 4
#define CCH   64
#define HH    128
#define WW    128
#define HWPX  (HH*WW)
static const size_t NB = (size_t)BATCH * CCH * HWPX; // 4,194,304 floats

__device__ __forceinline__ u16 f2bf(float f) {
    unsigned u = __float_as_uint(f);
    u += 0x7FFFu + ((u >> 16) & 1u);
    return (u16)(u >> 16);
}
__device__ __forceinline__ float bf2f(u16 h) {
    return __uint_as_float(((unsigned)h) << 16);
}
__device__ __forceinline__ float gelu_exact(float x) {
    return 0.5f * x * (1.0f + erff(x * 0.70710678118654752f));
}
// Low 32 bits of a generic pointer into the LDS aperture == wave-relative LDS
// byte address (ISA 10.2).
__device__ __forceinline__ unsigned lds_addr_of(const void* p) {
    return (unsigned)(size_t)p;
}
// Async copy 16B global -> LDS (ASYNCcnt-tracked), ISA GLOBAL_LOAD_ASYNC_TO_LDS_B128
__device__ __forceinline__ void async_g2l_b128(unsigned lds_byte, u64 gaddr) {
    asm volatile("global_load_async_to_lds_b128 %0, %1, off"
                 :: "v"(lds_byte), "v"(gaddr) : "memory");
}
__device__ __forceinline__ void wait_async0() {
#if __has_builtin(__builtin_amdgcn_s_wait_asynccnt)
    __builtin_amdgcn_s_wait_asynccnt(0);
#else
    asm volatile("s_wait_asynccnt 0x0" ::: "memory");
#endif
}

// ---------------------------------------------------------------------------
// Weight packing: f32 [Cout][Cin][kh][kw] -> bf16 A-fragment layout.
// K-order: k = (ky*3+kx)*Cin + ci  (ks9==9) or k = ci (ks9==1).
// 16-bit A layout (contiguous k order): g = lane>>4, M = lane&15,
//   K = 16g + j
// ---------------------------------------------------------------------------
__global__ void pack_weights(const float* __restrict__ w, u16* __restrict__ dst,
                             int Cin, int Cout, int ks9) {
    int total = Cout * Cin * ks9;
    int e = blockIdx.x * blockDim.x + threadIdx.x;
    if (e >= total) return;
    int numChunks = (Cin * ks9) / 32;
    int perMT = numChunks * 512;
    int mt = e / perMT;
    int r  = e - mt * perMT;
    int kc = r >> 9;
    int r2 = r & 511;
    int lane = r2 >> 4;
    int j    = r2 & 15;
    int g    = lane >> 4;
    int mrow = lane & 15;
    int kl = 16 * g + j;
    int k  = kc * 32 + kl;
    int co = mt * 16 + mrow;
    int widx;
    if (ks9 == 9) {
        int tap = k / Cin;
        int ci  = k - tap * Cin;
        widx = (co * Cin + ci) * 9 + tap;
    } else {
        widx = co * Cin + k;
    }
    dst[e] = f2bf(w[widx]);
}

// ---------------------------------------------------------------------------
// 3x3 conv (pad=1) as implicit GEMM via WMMA bf16.
// block = 128 thr (4 waves); wave -> 16 out-ch, block -> 16 px.
// grid = (8*HH, BATCH, CoutTot/64)
// INMODE: 0 = f32 NCHW (gather+convert), 2 = bf16 NHWC (async LDS copy)
// MODE:   0 plain f32, 1 relu f32, 2 +res f32 (NCHW),
//         3 bf16 NHWC store, 4 gelu-gate bf16 NHWC (aux bf16 NHWC)
// ---------------------------------------------------------------------------
template<int INMODE, int MODE, int CIN>
__global__ __launch_bounds__(128)
void conv3x3_wmma(const void* __restrict__ in, const u16* __restrict__ apack,
                  const float* __restrict__ bias, void* __restrict__ out,
                  const void* __restrict__ aux, int CoutTot) {
    extern __shared__ u16 sm[];                  // [3][18][CIN] bf16 patch
    const int tid = threadIdx.x;
    const int xt = blockIdx.x & 7;
    const int y  = blockIdx.x >> 3;
    const int b  = blockIdx.y;
    const int x0 = xt * 16;

    if (INMODE == 2) {
        // NHWC bf16 input: LDS patch layout == global layout -> async copy
        const unsigned smbase = lds_addr_of(sm);
        const u64 gbase = (u64)(size_t)in + (u64)b * HWPX * CIN * 2;
        const int chunks = 54 * CIN / 8;         // 16B chunks
        for (int q = tid; q < chunks; q += 128) {
            int pos = q / (CIN / 8);
            int cq  = q - pos * (CIN / 8);
            int dy = pos / 18, xi = pos - dy * 18;
            int gy = y + dy - 1, gx = x0 + xi - 1;
            unsigned loff = (unsigned)(pos * CIN + cq * 8) * 2;
            if (gy >= 0 && gy < HH && gx >= 0 && gx < WW) {
                u64 ga = gbase + ((u64)(gy * WW + gx) * CIN + cq * 8) * 2;
                async_g2l_b128(smbase + loff, ga);
            } else {
                u64* z = (u64*)((char*)sm + loff);
                z[0] = 0ull; z[1] = 0ull;
            }
        }
        wait_async0();
    } else {
        const int total = 54 * CIN;
        for (int e = tid; e < total; e += 128) {
            int dy = e / (18 * CIN);
            int r  = e - dy * 18 * CIN;
            int xi = r / CIN;
            int ci = r - xi * CIN;
            int gy = y + dy - 1, gx = x0 + xi - 1;
            u16 hv = 0;
            if (gy >= 0 && gy < HH && gx >= 0 && gx < WW) {
                size_t idx = ((size_t)(b * CIN + ci)) * HWPX + (size_t)gy * WW + gx;
                hv = f2bf(((const float*)in)[idx]);
            }
            sm[(dy * 18 + xi) * CIN + ci] = hv;
        }
    }
    __syncthreads();

    const int wave = tid >> 5, lane = tid & 31;
    const int g = lane >> 4, n = lane & 15;
    const int mt = blockIdx.z * 4 + wave;
    constexpr int numChunks = (CIN * 9) / 32;
    constexpr int cpt = CIN / 32;                // power of two for all layers

    v8f acc = {};
    const u16* ap = apack + (size_t)mt * numChunks * 512 + lane * 16;
    for (int kc = 0; kc < numChunks; ++kc) {
        int tap = kc / cpt;
        int cb  = (kc - tap * cpt) * 32;
        int ky = tap / 3, kx = tap - ky * 3;
        v16bf afrag = *(const v16bf*)ap;  ap += 512;
        v16bf bfrag = *(const v16bf*)&sm[(ky * 18 + n + kx) * CIN + cb + g * 16];
        acc = __builtin_amdgcn_wmma_f32_16x16x32_bf16(
                  false, afrag, false, bfrag, (short)0, acc, false, false);
    }

    const int co0 = mt * 16;
    const int pix = y * WW + x0 + n;
#pragma unroll
    for (int r = 0; r < 8; ++r) {
        int m = (g ? 8 : 0) + r;
        int co = co0 + m;
        float val = acc[r] + bias[co];
        size_t oc = ((size_t)(b * CoutTot + co)) * HWPX + pix;           // NCHW
        size_t oh = ((size_t)b * HWPX + pix) * CoutTot + co;             // NHWC
        if (MODE == 0) ((float*)out)[oc] = val;
        if (MODE == 1) ((float*)out)[oc] = fmaxf(val, 0.0f);
        if (MODE == 2) ((float*)out)[oc] = val + ((const float*)aux)[oc];
        if (MODE == 3) ((u16*)out)[oh] = f2bf(val);
        if (MODE == 4) {
            float gate = bf2f(((const u16*)aux)[oh]);
            ((u16*)out)[oh] = f2bf(gelu_exact(gate) * val);
        }
    }
}

// ---------------------------------------------------------------------------
// 1x1 conv as GEMM via WMMA bf16. grid = (HWPX/16, BATCH, CoutTot/64)
// ---------------------------------------------------------------------------
template<int INMODE, int MODE, int CIN>
__global__ __launch_bounds__(128)
void conv1x1_wmma(const void* __restrict__ in, const u16* __restrict__ apack,
                  const float* __restrict__ bias, void* __restrict__ out,
                  const void* __restrict__ aux, int CoutTot) {
    extern __shared__ u16 sm[];                  // [16][CIN] bf16
    const int tid = threadIdx.x;
    const int b = blockIdx.y;
    const int n0 = blockIdx.x * 16;

    if (INMODE == 2) {
        // NHWC bf16: 16 px * CIN ch fully contiguous in global -> async copy
        const unsigned smbase = lds_addr_of(sm);
        const u64 gbase = (u64)(size_t)in + ((u64)b * HWPX + n0) * CIN * 2;
        const int chunks = 16 * CIN / 8;
        for (int q = tid; q < chunks; q += 128)
            async_g2l_b128(smbase + q * 16, gbase + (u64)q * 16);
        wait_async0();
    } else {
        const int total = 16 * CIN;
        for (int e = tid; e < total; e += 128) {
            int px = e / CIN;
            int ci = e - px * CIN;
            size_t idx = ((size_t)(b * CIN + ci)) * HWPX + n0 + px;
            sm[px * CIN + ci] = f2bf(((const float*)in)[idx]);
        }
    }
    __syncthreads();

    const int wave = tid >> 5, lane = tid & 31;
    const int g = lane >> 4, n = lane & 15;
    const int mt = blockIdx.z * 4 + wave;
    constexpr int numChunks = CIN / 32;

    v8f acc = {};
    const u16* ap = apack + (size_t)mt * numChunks * 512 + lane * 16;
#pragma unroll
    for (int kc = 0; kc < numChunks; ++kc) {
        v16bf afrag = *(const v16bf*)ap;  ap += 512;
        v16bf bfrag = *(const v16bf*)&sm[n * CIN + kc * 32 + g * 16];
        acc = __builtin_amdgcn_wmma_f32_16x16x32_bf16(
                  false, afrag, false, bfrag, (short)0, acc, false, false);
    }

    const int co0 = mt * 16;
    const int pix = n0 + n;
#pragma unroll
    for (int r = 0; r < 8; ++r) {
        int m = (g ? 8 : 0) + r;
        int co = co0 + m;
        float val = acc[r] + bias[co];
        size_t oc = ((size_t)(b * CoutTot + co)) * HWPX + pix;           // NCHW
        size_t oh = ((size_t)b * HWPX + pix) * CoutTot + co;             // NHWC
        if (MODE == 0) ((float*)out)[oc] = val;
        if (MODE == 1) ((float*)out)[oc] = fmaxf(val, 0.0f);
        if (MODE == 2) ((float*)out)[oc] = val + ((const float*)aux)[oc];
        if (MODE == 3) ((u16*)out)[oh] = f2bf(val);
        if (MODE == 4) {
            float gate = bf2f(((const u16*)aux)[oh]);
            ((u16*)out)[oh] = f2bf(gelu_exact(gate) * val);
        }
    }
}

// ---------------------------------------------------------------------------
// depthwise Laplacian with reflect padding
// ---------------------------------------------------------------------------
__global__ void highfreq_k(const float* __restrict__ x, float* __restrict__ o) {
    size_t idx = (size_t)blockIdx.x * blockDim.x + threadIdx.x;
    if (idx >= NB) return;
    int px = (int)(idx % HWPX);
    size_t bc = idx / HWPX;
    int yy = px / WW, xx = px - yy * WW;
    const float* p = x + bc * HWPX;
    int ym = (yy == 0) ? 1 : yy - 1;
    int yp = (yy == HH - 1) ? HH - 2 : yy + 1;
    int xm = (xx == 0) ? 1 : xx - 1;
    int xp = (xx == WW - 1) ? WW - 2 : xx + 1;
    o[idx] = p[ym * WW + xx] + p[yp * WW + xx] + p[yy * WW + xm] + p[yy * WW + xp]
             - 4.0f * p[yy * WW + xx];
}

// mean over HW of exp(0.5*clip(lv,-10,10)) -> U[b*128 + mo*64 + c]
__global__ void reduceU_k(const float* __restrict__ lv, float* __restrict__ U, int mo) {
    __shared__ float sd[256];
    int c = blockIdx.x, b = blockIdx.y, t = threadIdx.x;
    const float* p = lv + ((size_t)(b * CCH + c)) * HWPX;
    float s = 0.0f;
    for (int i = t; i < HWPX; i += 256) {
        float v = p[i];
        v = fminf(fmaxf(v, -10.0f), 10.0f);
        s += expf(0.5f * v);
    }
    sd[t] = s; __syncthreads();
    for (int k = 128; k > 0; k >>= 1) { if (t < k) sd[t] += sd[t + k]; __syncthreads(); }
    if (t == 0) U[b * 128 + mo * 64 + c] = sd[0] / (float)HWPX;
}

// SE channel attention: U[4][128] -> wch[4][128]
__global__ void se_k(const float* __restrict__ U,
                     const float* __restrict__ fc1w, const float* __restrict__ fc1b,
                     const float* __restrict__ fc2w, const float* __restrict__ fc2b,
                     float* __restrict__ wch) {
    __shared__ float sU[512], sR[512], sZ[32], sMean[4];
    int t = threadIdx.x;
    sU[t] = U[t];
    __syncthreads();
    if (t < 4) {
        float m = 0.0f;
        for (int j = 0; j < 128; ++j) m += sU[t * 128 + j];
        sMean[t] = m / 128.0f;
    }
    __syncthreads();
    {
        int b = t >> 7;
        sR[t] = expf(-5.0f * sU[t] / (sMean[b] + 1e-6f));
    }
    __syncthreads();
    if (t < 32) {
        int b = t >> 3, h = t & 7;
        float z = fc1b[h];
        for (int j = 0; j < 128; ++j) z += sR[b * 128 + j] * fc1w[h * 128 + j];
        sZ[t] = fmaxf(z, 0.0f);
    }
    __syncthreads();
    {
        int b = t >> 7, j = t & 127;
        float s = fc2b[j];
        for (int h = 0; h < 8; ++h) s += sZ[b * 8 + h] * fc2w[j * 8 + h];
        wch[t] = 1.0f / (1.0f + expf(-s));
    }
}

// X_enh = X * (1 + wch)  (X = concat(F_ct, F_mr) on channel)
__global__ void xenh_k(const float* __restrict__ fct, const float* __restrict__ fmr,
                       const float* __restrict__ wch, float* __restrict__ o) {
    size_t idx = (size_t)blockIdx.x * blockDim.x + threadIdx.x;
    if (idx >= 2 * NB) return;
    int n = (int)(idx % HWPX);
    size_t bc = idx / HWPX;
    int cc = (int)(bc % 128);
    int b = (int)(bc / 128);
    float v = (cc < 64) ? fct[((size_t)(b * 64 + cc)) * HWPX + n]
                        : fmr[((size_t)(b * 64 + cc - 64)) * HWPX + n];
    o[idx] = v * (1.0f + wch[b * 128 + cc]);
}

// LayerNorm over HW: stats then apply
__global__ void ln_stats_k(const float* __restrict__ x, float* __restrict__ muv) {
    __shared__ float ss[256], sq[256];
    int c = blockIdx.x, b = blockIdx.y, t = threadIdx.x;
    const float* p = x + ((size_t)(b * CCH + c)) * HWPX;
    float s = 0.0f, q = 0.0f;
    for (int i = t; i < HWPX; i += 256) { float v = p[i]; s += v; q += v * v; }
    ss[t] = s; sq[t] = q; __syncthreads();
    for (int k = 128; k > 0; k >>= 1) {
        if (t < k) { ss[t] += ss[t + k]; sq[t] += sq[t + k]; }
        __syncthreads();
    }
    if (t == 0) {
        float mu = ss[0] / (float)HWPX;
        float var = sq[0] / (float)HWPX - mu * mu;
        muv[(b * CCH + c) * 2 + 0] = mu;
        muv[(b * CCH + c) * 2 + 1] = rsqrtf(var + 1e-5f);
    }
}
__global__ void ln_apply_k(const float* __restrict__ x, const float* __restrict__ muv,
                           float* __restrict__ o) {
    size_t idx = (size_t)blockIdx.x * blockDim.x + threadIdx.x;
    if (idx >= NB) return;
    size_t bc = idx / HWPX;
    float mu = muv[bc * 2], rs = muv[bc * 2 + 1];
    o[idx] = (x[idx] - mu) * rs;
}

// A[b,c,d] = sum_n K[b,c,n]*Q[b,d,n]   (one block per batch, LDS tiled)
__global__ __launch_bounds__(256)
void attnA_k(const float* __restrict__ K, const float* __restrict__ Q,
             float* __restrict__ A) {
    __shared__ float lk[64 * 64], lq[64 * 64];
    int b = blockIdx.x, t = threadIdx.x;
    float acc[16];
#pragma unroll
    for (int i = 0; i < 16; ++i) acc[i] = 0.0f;
    for (int nt = 0; nt < HWPX / 64; ++nt) {
        for (int i = 0; i < 16; ++i) {
            int e = t + 256 * i;
            int c = e >> 6, nn = e & 63;
            lk[e] = K[((size_t)(b * 64 + c)) * HWPX + nt * 64 + nn];
            lq[e] = Q[((size_t)(b * 64 + c)) * HWPX + nt * 64 + nn];
        }
        __syncthreads();
#pragma unroll
        for (int i = 0; i < 16; ++i) {
            int p = t + 256 * i;
            int c = p >> 6, d = p & 63;
            float s = acc[i];
            for (int nn = 0; nn < 64; ++nn) s += lk[c * 64 + nn] * lq[d * 64 + nn];
            acc[i] = s;
        }
        __syncthreads();
    }
#pragma unroll
    for (int i = 0; i < 16; ++i) A[(size_t)b * 4096 + t + 256 * i] = acc[i];
}

// softmax over flattened 4096 per batch
__global__ void softmaxA_k(float* __restrict__ A) {
    __shared__ float sd[256];
    int b = blockIdx.x, t = threadIdx.x;
    float* p = A + (size_t)b * 4096;
    float mx = -1e30f;
    for (int i = t; i < 4096; i += 256) mx = fmaxf(mx, p[i]);
    sd[t] = mx; __syncthreads();
    for (int k = 128; k > 0; k >>= 1) { if (t < k) sd[t] = fmaxf(sd[t], sd[t + k]); __syncthreads(); }
    mx = sd[0]; __syncthreads();
    float s = 0.0f;
    for (int i = t; i < 4096; i += 256) { float e = expf(p[i] - mx); p[i] = e; s += e; }
    sd[t] = s; __syncthreads();
    for (int k = 128; k > 0; k >>= 1) { if (t < k) sd[t] += sd[t + k]; __syncthreads(); }
    float inv = 1.0f / sd[0];
    for (int i = t; i < 4096; i += 256) p[i] *= inv;
}

// out[b,d,n] = sum_c V[b,c,n] * A[b,c,d]
__global__ __launch_bounds__(256)
void attnOut_k(const float* __restrict__ V, const float* __restrict__ A,
               float* __restrict__ O) {
    __shared__ float la[4096], lv[64 * 64];
    int b = blockIdx.y, t = threadIdx.x;
    int n0 = blockIdx.x * 64;
    for (int i = 0; i < 16; ++i) la[t + 256 * i] = A[(size_t)b * 4096 + t + 256 * i];
    for (int i = 0; i < 16; ++i) {
        int e = t + 256 * i;
        int c = e >> 6, px = e & 63;
        lv[e] = V[((size_t)(b * 64 + c)) * HWPX + n0 + px];
    }
    __syncthreads();
#pragma unroll
    for (int i = 0; i < 16; ++i) {
        int p = t + 256 * i;
        int d = p >> 6, px = p & 63;
        float s = 0.0f;
        for (int c = 0; c < 64; ++c) s += lv[c * 64 + px] * la[c * 64 + d];
        O[((size_t)(b * 64 + d)) * HWPX + n0 + px] = s;
    }
}

// ---------------------------------------------------------------------------
extern "C" void kernel_launch(void* const* d_in, const int* in_sizes, int n_in,
                              void* d_out, int out_size, void* d_ws, size_t ws_size,
                              hipStream_t stream) {
    (void)in_sizes; (void)n_in; (void)out_size; (void)ws_size;
    const float* F_ct = (const float*)d_in[0];
    const float* F_mr = (const float*)d_in[1];
    const float* uw1[2] = { (const float*)d_in[2], (const float*)d_in[6] };
    const float* ub1[2] = { (const float*)d_in[3], (const float*)d_in[7] };
    const float* uw2[2] = { (const float*)d_in[4], (const float*)d_in[8] };
    const float* ub2[2] = { (const float*)d_in[5], (const float*)d_in[9] };
    const float* fc1w = (const float*)d_in[10];
    const float* fc1b = (const float*)d_in[11];
    const float* fc2w = (const float*)d_in[12];
    const float* fc2b = (const float*)d_in[13];
    const float* mlpw = (const float*)d_in[14];
    const float* mlpb = (const float*)d_in[15];
    const float* qkvw1[3] = { (const float*)d_in[16], (const float*)d_in[20], (const float*)d_in[24] };
    const float* qkvb1[3] = { (const float*)d_in[17], (const float*)d_in[21], (const float*)d_in[25] };
    const float* qkvw2[3] = { (const float*)d_in[18], (const float*)d_in[22], (const float*)d_in[26] };
    const float* qkvb2[3] = { (const float*)d_in[19], (const float*)d_in[23], (const float*)d_in[27] };
    const float* w4 = (const float*)d_in[28];
    const float* b4 = (const float*)d_in[29];
    const float* d1w1 = (const float*)d_in[30]; const float* d1b1 = (const float*)d_in[31];
    const float* d1w2 = (const float*)d_in[32]; const float* d1b2 = (const float*)d_in[33];
    const float* d2w1 = (const float*)d_in[34]; const float* d2b1 = (const float*)d_in[35];
    const float* d2w2 = (const float*)d_in[36]; const float* d2b2 = (const float*)d_in[37];
    const float* wow = (const float*)d_in[38]; const float* wob = (const float*)d_in[39];

    float* ws = (float*)d_ws;
    float* HF   = ws + 0 * NB;      // later Q
    float* H1   = ws + 1 * NB;      // later K
    float* LV   = ws + 2 * NB;      // later V
    float* XENH = ws + 3 * NB;      // 2NB; later T1 (bf16 NHWC, 4NB elems)
    float* X0   = ws + 5 * NB;      // mlp out; becomes X2 in-place
    float* XN   = ws + 6 * NB;      // layernorm output (mdta + gdfn)
    float* TMP  = ws + 7 * NB;      // qkv 1x1 out; with ATT region -> T2 bf16
    float* Qb   = HF; float* Kb = H1; float* Vb = LV;
    float* ATT  = ws + 8 * NB;
    float* SMALL = ws + 9 * NB;
    float* U    = SMALL;            // 512
    float* WCH  = SMALL + 512;      // 512
    float* MUV  = SMALL + 1024;     // 512
    float* AMAT = SMALL + 2048;     // 16384
    u16*   PACK = (u16*)(SMALL + 2048 + 16384);
    u16*   T1   = (u16*)XENH;       // [B][HW][256] bf16 NHWC
    u16*   T2   = (u16*)TMP;        // [B][HW][256] bf16 NHWC

    size_t po = 0;
    auto PK = [&](int Cout, int K) { size_t o = po; po += (size_t)Cout * K; return o; };
    size_t p_uw1[2], p_uw2[2], p_qw1[3], p_qw2[3];
    p_uw1[0] = PK(64, 576);  p_uw2[0] = PK(64, 576);
    p_uw1[1] = PK(64, 576);  p_uw2[1] = PK(64, 576);
    size_t p_mlp = PK(64, 128);
    for (int i = 0; i < 3; ++i) { p_qw1[i] = PK(64, 64); p_qw2[i] = PK(64, 576); }
    size_t p_w4   = PK(64, 64);
    size_t p_d1w1 = PK(256, 64);  size_t p_d1w2 = PK(256, 2304);
    size_t p_d2w1 = PK(256, 64);  size_t p_d2w2 = PK(256, 2304);
    size_t p_wo   = PK(64, 256);

    auto packL = [&](const float* w, size_t off, int Cin, int Cout, int ks9) {
        int total = Cout * Cin * ks9;
        pack_weights<<<(total + 255) / 256, 256, 0, stream>>>(w, PACK + off, Cin, Cout, ks9);
    };
    for (int m = 0; m < 2; ++m) {
        packL(uw1[m], p_uw1[m], 64, 64, 9);
        packL(uw2[m] + (size_t)64 * 64 * 9, p_uw2[m], 64, 64, 9);  // log_var half only
    }
    packL(mlpw, p_mlp, 128, 64, 1);
    for (int i = 0; i < 3; ++i) { packL(qkvw1[i], p_qw1[i], 64, 64, 1); packL(qkvw2[i], p_qw2[i], 64, 64, 9); }
    packL(w4, p_w4, 64, 64, 1);
    packL(d1w1, p_d1w1, 64, 256, 1);  packL(d1w2, p_d1w2, 256, 256, 9);
    packL(d2w1, p_d2w1, 64, 256, 1);  packL(d2w2, p_d2w2, 256, 256, 9);
    packL(wow, p_wo, 256, 64, 1);

    const dim3 g3(8 * HH, BATCH, 1), g3b(8 * HH, BATCH, 4);
    const dim3 g1(HWPX / 16, BATCH, 1), g1b(HWPX / 16, BATCH, 4);
    const size_t sh64 = 54 * 64 * sizeof(u16), sh256 = 54 * 256 * sizeof(u16);
    const size_t s164 = 16 * 64 * sizeof(u16), s1128 = 16 * 128 * sizeof(u16), s1256 = 16 * 256 * sizeof(u16);
    const int EB = 256;
    const size_t nbBlocks = (NB + EB - 1) / EB;

    // ---- frequency fusion ----
    const float* FIN[2] = { F_ct, F_mr };
    for (int m = 0; m < 2; ++m) {
        highfreq_k<<<nbBlocks, EB, 0, stream>>>(FIN[m], HF);
        conv3x3_wmma<0, 1, 64><<<g3, 128, sh64, stream>>>(HF, PACK + p_uw1[m], ub1[m], H1, nullptr, 64);
        conv3x3_wmma<0, 0, 64><<<g3, 128, sh64, stream>>>(H1, PACK + p_uw2[m], ub2[m] + 64, LV, nullptr, 64);
        reduceU_k<<<dim3(64, BATCH), 256, 0, stream>>>(LV, U, m);
    }
    se_k<<<1, 512, 0, stream>>>(U, fc1w, fc1b, fc2w, fc2b, WCH);
    xenh_k<<<(2 * NB + EB - 1) / EB, EB, 0, stream>>>(F_ct, F_mr, WCH, XENH);

    // ---- restormer: mlp 1x1 (128->64) ----
    conv1x1_wmma<0, 0, 128><<<g1, 128, s1128, stream>>>(XENH, PACK + p_mlp, mlpb, X0, nullptr, 64);

    // ---- MDTA ----
    ln_stats_k<<<dim3(64, BATCH), 256, 0, stream>>>(X0, MUV);
    ln_apply_k<<<nbBlocks, EB, 0, stream>>>(X0, MUV, XN);
    float* QKV[3] = { Qb, Kb, Vb };
    for (int i = 0; i < 3; ++i) {
        conv1x1_wmma<0, 0, 64><<<g1, 128, s164, stream>>>(XN, PACK + p_qw1[i], qkvb1[i], TMP, nullptr, 64);
        conv3x3_wmma<0, 0, 64><<<g3, 128, sh64, stream>>>(TMP, PACK + p_qw2[i], qkvb2[i], QKV[i], nullptr, 64);
    }
    attnA_k<<<BATCH, 256, 0, stream>>>(Kb, Qb, AMAT);
    softmaxA_k<<<BATCH, 256, 0, stream>>>(AMAT);
    attnOut_k<<<dim3(HWPX / 64, BATCH), 256, 0, stream>>>(Vb, AMAT, ATT);
    conv1x1_wmma<0, 2, 64><<<g1, 128, s164, stream>>>(ATT, PACK + p_w4, b4, X0, X0, 64); // X2 in-place

    // ---- GDFN (bf16 NHWC intermediates, async-LDS staged WMMA) ----
    ln_stats_k<<<dim3(64, BATCH), 256, 0, stream>>>(X0, MUV);
    ln_apply_k<<<nbBlocks, EB, 0, stream>>>(X0, MUV, XN);
    conv1x1_wmma<0, 3, 64><<<g1b, 128, s164, stream>>>(XN, PACK + p_d1w1, d1b1, T1, nullptr, 256);
    conv3x3_wmma<2, 3, 256><<<g3b, 128, sh256, stream>>>(T1, PACK + p_d1w2, d1b2, T2, nullptr, 256);
    conv1x1_wmma<0, 3, 64><<<g1b, 128, s164, stream>>>(XN, PACK + p_d2w1, d2b1, T1, nullptr, 256);
    conv3x3_wmma<2, 4, 256><<<g3b, 128, sh256, stream>>>(T1, PACK + p_d2w2, d2b2, T2, T2, 256); // g=gelu(t2)*t2'
    conv1x1_wmma<2, 2, 256><<<g1, 128, s1256, stream>>>(T2, PACK + p_wo, wob, d_out, X0, 64);   // + residual
}
